// TransformerEncoders2_34883724378219
// MI455X (gfx1250) — matrix operand order repaired and checked by reading, not compiled, not run on hardware
//
#include <hip/hip_runtime.h>
#include <hip/hip_bf16.h>

// ---------------------------------------------------------------------------
// TransformerEncoders2 for MI455X (gfx1250, wave32, WMMA + async-to-LDS)
//
// Input pointer layout assumption (setup_inputs() insertion order top-level;
// jax pytree (sorted-key) order inside 'params'):
//   d_in[0..8]   : x_pf,x_f,x_lt,x_c,x_rt,x_lp,x_p,x_rp,x_o   [B,S,d] f32
//   d_in[9..12]  : bn_b, bn_g, bn_m, bn_v                      [10] f32
//   d_in[13+12e+k], e=0..8, k: W1,W2,Wo,Wqkv,b1,b2,bf1,bf2,bo,bqkv,g1,g2
//   d_in[121]    : lin_W [10,310]     d_in[122] : lin_b [10]
//
// All WMMA operands are pre-packed zero-padded f16 so every fragment load is
// an unconditional aligned b128 (global, async-to-LDS staged, or ds).  GEMM
// outputs are column-padded to the 16-wide tile so stores are unguarded.
// ---------------------------------------------------------------------------

typedef __attribute__((ext_vector_type(16))) _Float16 v16h;
typedef __attribute__((ext_vector_type(8)))  _Float16 h8;
typedef __attribute__((ext_vector_type(8)))  float    v8f;

#define BATCH 64
#define SEQ   512
#define MTOK  (BATCH * SEQ)
#define NHEADS 5
#define EPSLN 1e-5f
#define QTPB  8   // query tiles (waves) per attention block

union HFrag { v16h v; h8 h[2]; _Float16 e[16]; };

__device__ __forceinline__ v8f wmma_f16(v16h a, v16h b, v8f c) {
  return __builtin_amdgcn_wmma_f32_16x16x32_f16(false, a, false, b,
                                                (short)0, c, false, false);
}

// Async copy of one 16-byte chunk global -> LDS (per-lane), ASYNCcnt-tracked.
__device__ __forceinline__ void async_b128(unsigned lds_off, const void* gptr) {
  asm volatile("global_load_async_to_lds_b128 %0, %1, off"
               :: "v"(lds_off), "v"(gptr) : "memory");
}
__device__ __forceinline__ void wait_async0() {
  asm volatile("s_wait_asynccnt 0x0" ::: "memory");
}

// ---------------------------------------------------------------------------
// Pack f32 [R x K] -> f16 [Rp x Kp], zero padded (Kp mult of 32, Rp of 16).
// src row stride = ldsrc.
// ---------------------------------------------------------------------------
__global__ void pack_f16(const float* __restrict__ src, _Float16* __restrict__ dst,
                         int R, int K, int ldsrc, int Kp, int total) {
  const int i = blockIdx.x * blockDim.x + threadIdx.x;
  if (i >= total) return;
  const int k = i % Kp, r = i / Kp;
  dst[i] = (r < R && k < K) ? (_Float16)src[(size_t)r * ldsrc + k]
                            : (_Float16)0.f;
}

// Zero-pad a bias vector into f32 [Np].
__global__ void pack_bias(const float* __restrict__ b, float* __restrict__ bp,
                          int N, int Np) {
  const int i = blockIdx.x * blockDim.x + threadIdx.x;
  if (i < Np) bp[i] = (i < N) ? b[i] : 0.f;
}

// Q/K pack: qkv [M x ldq] f32 -> Qh/Kh [B,H,S,32] f16 (head dim padded to 32,
// Q pre-scaled by 1/sqrt(dh)).
__global__ void pack_qk(const float* __restrict__ qkv, _Float16* __restrict__ Qh,
                        _Float16* __restrict__ Kh, int d, int dh, int ldq,
                        float scale) {
  const int i = blockIdx.x * blockDim.x + threadIdx.x;
  if (i >= BATCH * NHEADS * SEQ * 32) return;
  const int c = i & 31;
  const int t = i >> 5;
  const int s = t & (SEQ - 1);
  const int bh = t >> 9;
  const int h = bh % NHEADS, b = bh / NHEADS;
  const size_t rowq = (size_t)(b * SEQ + s) * ldq;
  float q = 0.f, k = 0.f;
  if (c < dh) {
    q = qkv[rowq + h * dh + c] * scale;
    k = qkv[rowq + d + h * dh + c];
  }
  Qh[i] = (_Float16)q;
  Kh[i] = (_Float16)k;
}

// V pack (transposed): qkv -> Vt [B,H,16,S] f16 (head dim padded to 16).
__global__ void pack_v(const float* __restrict__ qkv, _Float16* __restrict__ Vt,
                       int d, int dh, int ldq) {
  const int i = blockIdx.x * blockDim.x + threadIdx.x;
  if (i >= BATCH * NHEADS * 16 * SEQ) return;
  const int s = i & (SEQ - 1);
  const int t = i >> 9;
  const int c = t & 15;
  const int bh = t >> 4;
  const int h = bh % NHEADS, b = bh / NHEADS;
  float v = 0.f;
  if (c < dh) v = qkv[(size_t)(b * SEQ + s) * ldq + 2 * d + h * dh + c];
  Vt[i] = (_Float16)v;
}

// ---------------------------------------------------------------------------
// C[M x Np] = Ah[M x Kp] * Wh[Np x Kp]^T + biasp, optional ReLU. Kp in {32,64}.
// One wave per 16-row stripe; A fragments live in registers across all
// N-tiles; weight fragments are software-pipelined one tile ahead; bias and
// output are tile-padded so there is no EXEC divergence anywhere.
// ---------------------------------------------------------------------------
__global__ __launch_bounds__(32)
void wmma_gemm_bias(const _Float16* __restrict__ Ah,
                    const _Float16* __restrict__ Wh,
                    const float* __restrict__ biasp, float* __restrict__ C,
                    int Kp, int ldc, int relu, int ntiles) {
  const int mt = blockIdx.x;
  const int lane = threadIdx.x & 31;
  const int half = lane >> 4, lo = lane & 15;
  const _Float16* arow = Ah + (size_t)(mt * 16 + lo) * Kp;
  const bool k64 = (Kp > 32);

  HFrag a0, a1, b0, b1;
#pragma unroll
  for (int j = 0; j < 16; ++j) { a1.e[j] = (_Float16)0.f; b1.e[j] = (_Float16)0.f; }

  a0.h[0] = *(const h8*)(arow + 16 * half);
  a0.h[1] = *(const h8*)(arow + 16 * half + 8);
  if (k64) {
    a1.h[0] = *(const h8*)(arow + 32 + 16 * half);
    a1.h[1] = *(const h8*)(arow + 32 + 16 * half + 8);
  }

  {
    const _Float16* w = Wh + (size_t)lo * Kp;
    b0.h[0] = *(const h8*)(w + 16 * half);
    b0.h[1] = *(const h8*)(w + 16 * half + 8);
    if (k64) {
      b1.h[0] = *(const h8*)(w + 32 + 16 * half);
      b1.h[1] = *(const h8*)(w + 32 + 16 * half + 8);
    }
  }

  for (int nt = 0; nt < ntiles; ++nt) {
    HFrag c0 = b0, c1 = b1;
    if (nt + 1 < ntiles) {   // prefetch next tile's weights during the WMMAs
      const _Float16* w = Wh + (size_t)((nt + 1) * 16 + lo) * Kp;
      b0.h[0] = *(const h8*)(w + 16 * half);
      b0.h[1] = *(const h8*)(w + 16 * half + 8);
      if (k64) {
        b1.h[0] = *(const h8*)(w + 32 + 16 * half);
        b1.h[1] = *(const h8*)(w + 32 + 16 * half + 8);
      }
    }

    const int col = nt * 16 + lo;
    v8f acc;
    const float bv = biasp[col];
#pragma unroll
    for (int r = 0; r < 8; ++r) acc[r] = bv;

    acc = wmma_f16(a0.v, c0.v, acc);
    if (k64) acc = wmma_f16(a1.v, c1.v, acc);

#pragma unroll
    for (int r = 0; r < 8; ++r) {
      float v = acc[r];
      if (relu) v = fmaxf(v, 0.f);
      C[(size_t)(mt * 16 + r + 8 * half) * ldc + col] = v;
    }
  }
}

// ---------------------------------------------------------------------------
// Flash attention: 8 waves per block, one (b, h, group of 8 query tiles).
// K ([S,32] f16, 32KB) and Vt ([16,S] f16, 16KB) are staged once into LDS
// with global_load_async_to_lds_b128; waves then stream from LDS with
// ds_load_b128.  Qh/Kh: [B,H,S,32] (Q pre-scaled); Vt: [B,H,16,S].
// ---------------------------------------------------------------------------
__global__ __launch_bounds__(256)
void attn_flash(const _Float16* __restrict__ Qh, const _Float16* __restrict__ Kh,
                const _Float16* __restrict__ Vt, float* __restrict__ ctx,
                int d, int dh) {
  const int qg = blockIdx.x, h = blockIdx.y, b = blockIdx.z;
  const int tid = threadIdx.x;
  const int wid = tid >> 5;
  const int lane = tid & 31;
  const int half = lane >> 4, lo = lane & 15;
  const int qt = qg * QTPB + wid;

  __shared__ _Float16 kbuf[SEQ * 32];        // 32 KB
  __shared__ _Float16 vbuf[16 * SEQ];        // 16 KB
  __shared__ _Float16 pbuf[QTPB][16 * 32];   //  8 KB (per-wave P tiles)

  const size_t bh = (size_t)(b * NHEADS + h);
  const _Float16* kg = Kh + bh * SEQ * 32;
  const _Float16* vg = Vt + bh * 16 * SEQ;

  // --- Q fragment (A layout) from global: overlaps the LDS staging ---
  HFrag qa;
  {
    const _Float16* qp = Qh + (bh * SEQ + qt * 16 + lo) * 32;
    qa.h[0] = *(const h8*)(qp + 16 * half);
    qa.h[1] = *(const h8*)(qp + 16 * half + 8);
  }

  // --- Stage K and V into LDS (async, ASYNCcnt) ---
  {
    const unsigned kofs = (unsigned)(uintptr_t)kbuf;
    const unsigned vofs = (unsigned)(uintptr_t)vbuf;
#pragma unroll
    for (int it = 0; it < 8; ++it) {          // 2048 b128 chunks of K
      const int idx = tid + it * 256;
      async_b128(kofs + idx * 16, (const void*)(kg + idx * 8));
    }
#pragma unroll
    for (int it = 0; it < 4; ++it) {          // 1024 b128 chunks of V
      const int idx = tid + it * 256;
      async_b128(vofs + idx * 16, (const void*)(vg + idx * 8));
    }
    wait_async0();
  }
  __syncthreads();

  float m_run[8], l_run[8];
  v8f o;
#pragma unroll
  for (int r = 0; r < 8; ++r) { m_run[r] = -1e30f; l_run[r] = 0.f; o[r] = 0.f; }

  _Float16* pw = pbuf[wid];

  for (int kc = 0; kc < SEQ; kc += 32) {
    // Two 16-key score tiles from LDS (B-layout: lane = key, elems = dim).
    HFrag kb0, kb1;
    {
      const _Float16* kp0 = kbuf + (kc + lo) * 32 + 16 * half;
      const _Float16* kp1 = kbuf + (kc + 16 + lo) * 32 + 16 * half;
      kb0.h[0] = *(const h8*)(kp0);
      kb0.h[1] = *(const h8*)(kp0 + 8);
      kb1.h[0] = *(const h8*)(kp1);
      kb1.h[1] = *(const h8*)(kp1 + 8);
    }
    v8f s0 = {}, s1 = {};
    s0 = wmma_f16(qa.v, kb0.v, s0);
    s1 = wmma_f16(qa.v, kb1.v, s1);

    // Online softmax per row r (row = r + 8*half, 16 lanes = 16 keys/tile).
#pragma unroll
    for (int r = 0; r < 8; ++r) {
      const float v0 = s0[r], v1 = s1[r];
      float mx = fmaxf(v0, v1);
      mx = fmaxf(mx, __shfl_xor(mx, 1, 32));
      mx = fmaxf(mx, __shfl_xor(mx, 2, 32));
      mx = fmaxf(mx, __shfl_xor(mx, 4, 32));
      mx = fmaxf(mx, __shfl_xor(mx, 8, 32));
      const float m_new = fmaxf(m_run[r], mx);
      const float p0 = __expf(v0 - m_new);
      const float p1 = __expf(v1 - m_new);
      float ps = p0 + p1;
      ps += __shfl_xor(ps, 1, 32);
      ps += __shfl_xor(ps, 2, 32);
      ps += __shfl_xor(ps, 4, 32);
      ps += __shfl_xor(ps, 8, 32);
      const float alpha = __expf(m_run[r] - m_new);
      l_run[r] = l_run[r] * alpha + ps;
      m_run[r] = m_new;
      o[r] *= alpha;
      const int row = r + 8 * half;
      pw[row * 32 + lo] = (_Float16)p0;
      pw[row * 32 + 16 + lo] = (_Float16)p1;
    }
    // Per-wave LDS ordering is in-order; no block barrier needed.

    // PV: A = P (16x32, full K), B = V chunk (32 keys x 16 dims) from LDS.
    HFrag pa, vb;
    pa.h[0] = *(const h8*)(pw + lo * 32 + 16 * half);
    pa.h[1] = *(const h8*)(pw + lo * 32 + 16 * half + 8);
    {
      const _Float16* vp = vbuf + lo * SEQ + kc + 16 * half;
      vb.h[0] = *(const h8*)(vp);
      vb.h[1] = *(const h8*)(vp + 8);
    }
    o = wmma_f16(pa.v, vb.v, o);
  }

  if (lo < dh) {
#pragma unroll
    for (int r = 0; r < 8; ++r) {
      const int m = b * SEQ + qt * 16 + r + 8 * half;
      ctx[(size_t)m * d + h * dh + lo] = o[r] / l_run[r];
    }
  }
}

// ---------------------------------------------------------------------------
// out[t, :] = LayerNorm(x[t, :] + y[t, :]) * g + b   (one thread per token)
// x stride d, y stride ldy, out stride ldo.
// ---------------------------------------------------------------------------
__global__ void residual_ln(const float* __restrict__ x,
                            const float* __restrict__ y, int ldy,
                            const float* __restrict__ g,
                            const float* __restrict__ bvec,
                            float* __restrict__ out, int ldo, int d) {
  const int t = blockIdx.x * blockDim.x + threadIdx.x;
  if (t >= MTOK) return;
  float buf[64];
  float mean = 0.f;
  for (int i = 0; i < d; ++i) {
    const float v = x[(size_t)t * d + i] + y[(size_t)t * ldy + i];
    buf[i] = v;
    mean += v;
  }
  mean /= (float)d;
  float var = 0.f;
  for (int i = 0; i < d; ++i) {
    const float c = buf[i] - mean;
    var += c * c;
  }
  var /= (float)d;
  const float inv = rsqrtf(var + EPSLN);
  for (int i = 0; i < d; ++i)
    out[(size_t)t * ldo + i] = (buf[i] - mean) * inv * g[i] + bvec[i];
}

// ---------------------------------------------------------------------------
// y = concat[M x 310] @ lin_W[10 x 310]^T + lin_b, then eval-mode BatchNorm.
// ---------------------------------------------------------------------------
__global__ void final_linear_bn(const float* __restrict__ xc,
                                const float* __restrict__ Wl,
                                const float* __restrict__ bl,
                                const float* __restrict__ bng,
                                const float* __restrict__ bnb,
                                const float* __restrict__ bnm,
                                const float* __restrict__ bnv,
                                float* __restrict__ out) {
  const int idx = blockIdx.x * blockDim.x + threadIdx.x;
  if (idx >= MTOK * 10) return;
  const int t = idx / 10, oc = idx % 10;
  float s = bl[oc];
  const float* xr = xc + (size_t)t * 310;
  const float* wr = Wl + (size_t)oc * 310;
  for (int k = 0; k < 310; ++k) s += xr[k] * wr[k];
  const float inv = rsqrtf(bnv[oc] + EPSLN);
  out[(size_t)t * 10 + oc] = (s - bnm[oc]) * inv * bng[oc] + bnb[oc];
}

// ---------------------------------------------------------------------------
extern "C" void kernel_launch(void* const* d_in, const int* in_sizes, int n_in,
                              void* d_out, int out_size, void* d_ws,
                              size_t ws_size, hipStream_t stream) {
  static const int DIMS[9] = {25, 45, 30, 50, 30, 30, 45, 30, 25};
  const int M = MTOK;
  (void)in_sizes; (void)n_in; (void)out_size; (void)ws_size;

  const float* bn_b = (const float*)d_in[9];
  const float* bn_g = (const float*)d_in[10];
  const float* bn_m = (const float*)d_in[11];
  const float* bn_v = (const float*)d_in[12];
  const float* lin_W = (const float*)d_in[13 + 9 * 12];
  const float* lin_b = (const float*)d_in[13 + 9 * 12 + 1];

  // Workspace layout (float units; all segment sizes multiples of 4 so the
  // f16 segments stay 16-byte aligned).  GEMM outputs are column-padded.
  float* ws = (float*)d_ws;
  float* concat = ws;                               // M*310 f32
  float* qkvb   = concat + (size_t)M * 310;         // M*160 f32 (ld Np3)
  float* ctx    = qkvb   + (size_t)M * 160;         // M*50  f32
  float* x1     = ctx    + (size_t)M * 50;          // M*50  f32
  float* tmp    = x1     + (size_t)M * 50;          // M*64  f32 (ld Npd/64)
  float* ffb    = tmp    + (size_t)M * 64;          // M*64  f32
  float* Bp     = ffb    + (size_t)M * 64;          // 256 f32 padded bias
  _Float16* Ah  = (_Float16*)(Bp + 256);                    // M*64 f16
  _Float16* Wh  = (_Float16*)((float*)Ah + (size_t)M * 32); // 160*64 f16
  _Float16* Qh  = (_Float16*)((float*)Wh + 8192);           // B*H*S*32 f16
  _Float16* Kh  = (_Float16*)((float*)Qh + (size_t)BATCH * NHEADS * SEQ * 16);
  _Float16* Vt  = (_Float16*)((float*)Kh + (size_t)BATCH * NHEADS * SEQ * 16);

  int col_off = 0;
  for (int e = 0; e < 9; ++e) {
    const int d = DIMS[e], dh = d / NHEADS;
    const int Kp = (d + 31) & ~31;          // padded K for d-sized inner dims
    const int Np3 = (3 * d + 15) & ~15;     // padded N for qkv output
    const int Npd = (d + 15) & ~15;         // padded N for d-sized outputs
    const float* xe = (const float*)d_in[e];
    const int base = 13 + e * 12;
    const float* W1   = (const float*)d_in[base + 0];
    const float* W2   = (const float*)d_in[base + 1];
    const float* Wo   = (const float*)d_in[base + 2];
    const float* Wqkv = (const float*)d_in[base + 3];
    const float* b1   = (const float*)d_in[base + 4];
    const float* b2   = (const float*)d_in[base + 5];
    const float* bf1  = (const float*)d_in[base + 6];
    const float* bf2  = (const float*)d_in[base + 7];
    const float* bo   = (const float*)d_in[base + 8];
    const float* bqkv = (const float*)d_in[base + 9];
    const float* g1   = (const float*)d_in[base + 10];
    const float* g2   = (const float*)d_in[base + 11];

    // qkv = x @ Wqkv^T + bqkv   [M x 3d], stored with ld = Np3
    pack_f16<<<(M * Kp + 255) / 256, 256, 0, stream>>>(xe, Ah, M, d, d, Kp,
                                                       M * Kp);
    pack_f16<<<(Np3 * Kp + 255) / 256, 256, 0, stream>>>(Wqkv, Wh, 3 * d, d, d,
                                                         Kp, Np3 * Kp);
    pack_bias<<<1, 256, 0, stream>>>(bqkv, Bp, 3 * d, Np3);
    wmma_gemm_bias<<<M / 16, 32, 0, stream>>>(Ah, Wh, Bp, qkvb, Kp, Np3, 0,
                                              Np3 / 16);

    // ctx = softmax(QK^T/sqrt(dh)) V   [M x d]
    pack_qk<<<(BATCH * NHEADS * SEQ * 32 + 255) / 256, 256, 0, stream>>>(
        qkvb, Qh, Kh, d, dh, Np3, rsqrtf((float)dh));
    pack_v<<<(BATCH * NHEADS * 16 * SEQ + 255) / 256, 256, 0, stream>>>(
        qkvb, Vt, d, dh, Np3);
    attn_flash<<<dim3(SEQ / 16 / QTPB, NHEADS, BATCH), 256, 0, stream>>>(
        Qh, Kh, Vt, ctx, d, dh);

    // tmp = ctx @ Wo^T + bo (ld Npd) ; x1 = LN(x + tmp)
    pack_f16<<<(M * Kp + 255) / 256, 256, 0, stream>>>(ctx, Ah, M, d, d, Kp,
                                                       M * Kp);
    pack_f16<<<(Npd * Kp + 255) / 256, 256, 0, stream>>>(Wo, Wh, d, d, d, Kp,
                                                         Npd * Kp);
    pack_bias<<<1, 256, 0, stream>>>(bo, Bp, d, Npd);
    wmma_gemm_bias<<<M / 16, 32, 0, stream>>>(Ah, Wh, Bp, tmp, Kp, Npd, 0,
                                              Npd / 16);
    residual_ln<<<(M + 255) / 256, 256, 0, stream>>>(xe, tmp, Npd, g1, b1, x1,
                                                     d, d);

    // ff = relu(x1 @ W1^T + bf1) [M x 64] ; tmp = ff @ W2^T + bf2 (ld Npd)
    pack_f16<<<(M * Kp + 255) / 256, 256, 0, stream>>>(x1, Ah, M, d, d, Kp,
                                                       M * Kp);
    pack_f16<<<(64 * Kp + 255) / 256, 256, 0, stream>>>(W1, Wh, 64, d, d, Kp,
                                                        64 * Kp);
    pack_bias<<<1, 256, 0, stream>>>(bf1, Bp, 64, 64);
    wmma_gemm_bias<<<M / 16, 32, 0, stream>>>(Ah, Wh, Bp, ffb, Kp, 64, 1, 4);
    pack_f16<<<(M * 64 + 255) / 256, 256, 0, stream>>>(ffb, Ah, M, 64, 64, 64,
                                                       M * 64);
    pack_f16<<<(Npd * 64 + 255) / 256, 256, 0, stream>>>(W2, Wh, d, 64, 64, 64,
                                                         Npd * 64);
    pack_bias<<<1, 256, 0, stream>>>(bf2, Bp, d, Npd);
    wmma_gemm_bias<<<M / 16, 32, 0, stream>>>(Ah, Wh, Bp, tmp, 64, Npd, 0,
                                              Npd / 16);

    // concat[:, off:off+d] = LN(x1 + tmp)
    residual_ln<<<(M + 255) / 256, 256, 0, stream>>>(x1, tmp, Npd, g2, b2,
                                                     concat + col_off, 310, d);
    col_off += d;
  }

  final_linear_bn<<<(M * 10 + 255) / 256, 256, 0, stream>>>(
      concat, lin_W, lin_b, bn_g, bn_b, bn_m, bn_v, (float*)d_out);
}
